// MedicalMoE_36816459661688
// MI455X (gfx1250) — matrix operand order repaired and checked by reading, not compiled, not run on hardware
//
#include <hip/hip_runtime.h>
#include <stdint.h>
#include <stddef.h>

// ---------------- problem constants ----------------
#define BB   16384   // batch (tokens)
#define DD   1024    // d_model
#define HH   4096    // expert hidden
#define EE   8       // experts
#define KSEL 2       // top-k
#define GHH  256     // gating hidden

// ---------------- types ----------------
typedef __bf16 v16bf __attribute__((ext_vector_type(16)));
typedef __bf16 v8bf  __attribute__((ext_vector_type(8)));
typedef float  v8f   __attribute__((ext_vector_type(8)));
typedef unsigned v4u __attribute__((ext_vector_type(4)));
typedef int      v8i __attribute__((ext_vector_type(8)));
typedef int      v4i __attribute__((ext_vector_type(4)));

#if defined(__has_builtin)
#if __has_builtin(__builtin_amdgcn_tensor_load_to_lds)
#define HAVE_TDM 1
#endif
#if __has_builtin(__builtin_amdgcn_global_load_async_to_lds_b128) && \
    __has_builtin(__builtin_amdgcn_s_wait_asynccnt)
#define HAVE_ASYNC 1
#endif
#endif
#ifndef HAVE_TDM
#define HAVE_TDM 0
#endif
#ifndef HAVE_ASYNC
#define HAVE_ASYNC 0
#endif

#define ASG __attribute__((address_space(1)))
#define ASL __attribute__((address_space(3)))

__device__ __forceinline__ __bf16 f2bf(float f) {
    unsigned u = __builtin_bit_cast(unsigned, f);
    u += 0x7FFFu + ((u >> 16) & 1u);           // round-to-nearest-even
    unsigned short s = (unsigned short)(u >> 16);
    return __builtin_bit_cast(__bf16, s);
}
__device__ __forceinline__ float bf2f(__bf16 b) {
    unsigned short s = __builtin_bit_cast(unsigned short, b);
    unsigned u = ((unsigned)s) << 16;
    return __builtin_bit_cast(float, u);
}
// build a v16bf WMMA fragment from two 16-byte chunks
__device__ __forceinline__ v16bf ld16(const __bf16* p0, const __bf16* p1) {
    v8bf lo = *(const v8bf*)p0;
    v8bf hi = *(const v8bf*)p1;
    v16bf r;
#pragma unroll
    for (int i = 0; i < 8; ++i) { r[i] = lo[i]; r[i + 8] = hi[i]; }
    return r;
}

#if HAVE_TDM
// 2D tile load global->LDS via Tensor Data Mover (D# per ISA cdna5 ch.8).
__device__ __forceinline__ void tdm_load_2d(const void* g, unsigned ldsoff,
                                            unsigned tileK, unsigned tileRows,
                                            unsigned strideElems, unsigned flags) {
    unsigned long long ga = (unsigned long long)(size_t)g;
    v4u g0 = { 1u,                                  // count=1 valid descriptor
               ldsoff,                              // lds_addr
               (unsigned)ga,                        // global_addr[31:0]
               (((unsigned)(ga >> 32)) & 0x01FFFFFFu) | (2u << 30) }; // addr[56:32] | type=2
    const unsigned TD = 1u << 20;                   // oversized tensor dims (no OOB clipping)
    v8i g1;
    g1[0] = (int)flags;                             // wg_mask=0 | data_size | pad cfg
    g1[1] = (int)((TD & 0xFFFFu) << 16);            // tensor_dim0[15:0] -> bits 63:48
    g1[2] = (int)(((TD >> 16) & 0xFFFFu) | ((TD & 0xFFFFu) << 16)); // dim0 hi | dim1 lo
    g1[3] = (int)(((TD >> 16) & 0xFFFFu) | (tileK << 16));          // dim1 hi | tile_dim0
    g1[4] = (int)(tileRows & 0xFFFFu);              // tile_dim1 (tile_dim2 = 0)
    g1[5] = (int)strideElems;                       // tensor_dim0_stride[31:0]
    g1[6] = 0;
    g1[7] = 0;
    v4i z4 = { 0, 0, 0, 0 };
#if __clang_major__ >= 23
    v8i z8 = { 0, 0, 0, 0, 0, 0, 0, 0 };
    __builtin_amdgcn_tensor_load_to_lds(g0, g1, z4, z4, z8, 0);
#else
    __builtin_amdgcn_tensor_load_to_lds(g0, g1, z4, z4, 0);
#endif
}
// gemm1 B rows 128B data + 32B pad (interval 32 dw -> code 4, amount 8 dw -> code 7)
#define TDM_FLAGS_G1 ((1u << 16) | (1u << 20) | (4u << 22) | (7u << 25))
// gemm2 B rows 256B data + 16B pad (interval 64 dw -> code 5, amount 4 dw -> code 3)
#define TDM_FLAGS_G2 ((1u << 16) | (1u << 20) | (5u << 22) | (3u << 25))
#endif

#if HAVE_ASYNC
__device__ __forceinline__ void async_b128(const __bf16* src, __bf16* dst) {
    __builtin_amdgcn_global_load_async_to_lds_b128((ASG v4i*)src, (ASL v4i*)dst, 0, 0);
}
#endif

// ---------------- K0: fp32 -> bf16 elementwise ----------------
__global__ void cvt_x_kernel(const float* __restrict__ x, __bf16* __restrict__ xb, int n4) {
    int i = blockIdx.x * blockDim.x + threadIdx.x;
    if (i >= n4) return;
    const float4 v = *(const float4*)(x + (size_t)i * 4);
    __bf16* d = xb + (size_t)i * 4;
    d[0] = f2bf(v.x); d[1] = f2bf(v.y); d[2] = f2bf(v.z); d[3] = f2bf(v.w);
}

// ---------------- K1-3: tiled transpose + convert (in[R][C] -> out[C][R] bf16) -------------
__global__ void transpose_cvt_kernel(const float* __restrict__ in, __bf16* __restrict__ out,
                                     int R, int C) {
    __shared__ float tile[32][33];
    const size_t bstride = (size_t)R * C;
    const float* inp = in + (size_t)blockIdx.z * bstride;
    __bf16* outp = out + (size_t)blockIdx.z * bstride;
    int c0 = blockIdx.x * 32, r0 = blockIdx.y * 32;
    int tx = threadIdx.x, ty = threadIdx.y;
#pragma unroll
    for (int i = 0; i < 4; ++i) {
        int r = r0 + ty + i * 8;
        tile[ty + i * 8][tx] = inp[(size_t)r * C + c0 + tx];
    }
    __syncthreads();
#pragma unroll
    for (int i = 0; i < 4; ++i) {
        int c = c0 + ty + i * 8;
        outp[(size_t)c * R + r0 + tx] = f2bf(tile[tx][ty + i * 8]);
    }
}

// ---------------- K4: gating GEMM1 via WMMA: gh = relu(x @ gw1 + gb1), bf16 out ------------
#define GP 1032   // padded LDS pitch (516 dw, stride 4 banks -> conflict-free)
__global__ __launch_bounds__(256) void gating_gemm1_kernel(
    const __bf16* __restrict__ xb, const __bf16* __restrict__ gw1t,
    const float* __restrict__ gb1, __bf16* __restrict__ gh) {
    extern __shared__ __align__(16) char smem[];
    __bf16* xs = (__bf16*)smem;                       // [32][GP]
    const int tid = threadIdx.x;
    const int rowbase = blockIdx.x * 32;
    {
        int i = tid >> 3, seg = tid & 7;
        uint4* dst = (uint4*)(xs + i * GP + seg * 128);
        const uint4* src = (const uint4*)(xb + (size_t)(rowbase + i) * DD + seg * 128);
#pragma unroll
        for (int j = 0; j < 16; ++j) dst[j] = src[j];
    }
    __syncthreads();
    const int w = tid >> 5, lane = tid & 31;
    const int lh = lane >> 4, l15 = lane & 15;
    v8f acc[2][2];
#pragma unroll
    for (int mt = 0; mt < 2; ++mt)
#pragma unroll
        for (int nt = 0; nt < 2; ++nt)
#pragma unroll
            for (int r = 0; r < 8; ++r) acc[mt][nt][r] = 0.f;
    for (int kk = 0; kk < DD; kk += 32) {
        v16bf a[2];
#pragma unroll
        for (int mt = 0; mt < 2; ++mt) {
            const __bf16* p = xs + (mt * 16 + l15) * GP + kk + lh * 8;
            a[mt] = ld16(p, p + 16);
        }
#pragma unroll
        for (int nt = 0; nt < 2; ++nt) {
            const __bf16* bp = gw1t + (size_t)(w * 32 + nt * 16 + l15) * DD + kk + lh * 8;
            v16bf bf = ld16(bp, bp + 16);
#pragma unroll
            for (int mt = 0; mt < 2; ++mt)
                acc[mt][nt] = __builtin_amdgcn_wmma_f32_16x16x32_bf16(
                    false, a[mt], false, bf, (short)0, acc[mt][nt], false, false);
        }
    }
#pragma unroll
    for (int nt = 0; nt < 2; ++nt) {
        int n = w * 32 + nt * 16 + l15;
        float b = gb1[n];
#pragma unroll
        for (int mt = 0; mt < 2; ++mt)
#pragma unroll
            for (int r = 0; r < 8; ++r) {
                int m = mt * 16 + lh * 8 + r;
                float v = acc[mt][nt][r] + b;
                gh[(size_t)(rowbase + m) * GHH + n] = f2bf(v > 0.f ? v : 0.f);
            }
    }
}

// ---------------- K5: gating GEMM2 + softmax + top2 ----------------
__global__ void gating2_kernel(const __bf16* __restrict__ gh, const float* __restrict__ gw2,
                               const float* __restrict__ gb2, int* __restrict__ topk,
                               float* __restrict__ wnorm) {
    const int tl = threadIdx.x >> 3, e = threadIdx.x & 7;
    const int t = blockIdx.x * 32 + tl;
    float acc = gb2[e];
    const __bf16* g = gh + (size_t)t * GHH;
    for (int k = 0; k < GHH; ++k) acc += bf2f(g[k]) * gw2[k * EE + e];
    __shared__ float lg[32][8];
    lg[tl][e] = acc;
    __syncthreads();
    if (e == 0) {
        float l[8];
#pragma unroll
        for (int i = 0; i < 8; ++i) l[i] = lg[tl][i];
        float m = l[0];
#pragma unroll
        for (int i = 1; i < 8; ++i) m = fmaxf(m, l[i]);
        float p[8];
#pragma unroll
        for (int i = 0; i < 8; ++i) p[i] = __expf(l[i] - m);
        int i0 = 0;
#pragma unroll
        for (int i = 1; i < 8; ++i) if (p[i] > p[i0]) i0 = i;
        int i1 = (i0 == 0) ? 1 : 0;
#pragma unroll
        for (int i = 0; i < 8; ++i) if (i != i0 && p[i] > p[i1]) i1 = i;
        float ws = p[i0] + p[i1];
        topk[t * 2] = i0; topk[t * 2 + 1] = i1;
        wnorm[t * 2] = p[i0] / ws; wnorm[t * 2 + 1] = p[i1] / ws;
    }
}

// ---------------- K6: deterministic dispatch (one block per expert) ----------------
__global__ void dispatch_kernel(const int* __restrict__ topk, const float* __restrict__ wnorm,
                                int* __restrict__ sidList, float* __restrict__ wList,
                                int* __restrict__ counts) {
    const int e = blockIdx.x;
    const int tid = threadIdx.x;
    __shared__ int sc[256];
    int base = 0;
    for (int chunk = 0; chunk < BB / 256; ++chunk) {
        int t = chunk * 256 + tid;
        int k = -1;
        int i0 = topk[t * 2], i1 = topk[t * 2 + 1];
        if (i0 == e) k = 0; else if (i1 == e) k = 1;
        int v = (k >= 0) ? 1 : 0;
        sc[tid] = v;
        __syncthreads();
        for (int off = 1; off < 256; off <<= 1) {
            int x = (tid >= off) ? sc[tid - off] : 0;
            __syncthreads();
            sc[tid] += x;
            __syncthreads();
        }
        if (v) {
            int pos = base + sc[tid] - 1;
            sidList[e * BB + pos] = 2 * t + k;
            wList[e * BB + pos] = wnorm[2 * t + k];
        }
        base += sc[255];
        __syncthreads();
    }
    if (tid == 0) counts[e] = base;
}

// ================= expert GEMM1: h[sid] = tanh(x @ W1 + b1), bf16 =================
// block: 256 thr (8 waves, 4M x 2N), M-tile 128, all N=4096, K-chunks of 64 via TDM.
#define AP1 1032
#define BP1 80
#define B1SZ (128 * BP1)
__global__ __launch_bounds__(256) void egemm1_kernel(
    const __bf16* __restrict__ xb, const __bf16* __restrict__ w1t,
    const float* __restrict__ eb1, const int* __restrict__ sidList,
    const int* __restrict__ counts, __bf16* __restrict__ hbuf) {
    extern __shared__ __align__(16) char smem[];
    __bf16* As = (__bf16*)smem;                 // 128*1032 elems
    __bf16* Bs = As + 128 * AP1;                // 2*128*80 elems
    int* sidRow = (int*)(Bs + 2 * B1SZ);        // 128 ints
    const int e = blockIdx.y;
    const int count = counts[e];
    const int rowbase = blockIdx.x * 128;
    if (rowbase >= count) return;
    const int tid = threadIdx.x;
    if (tid < 128) {
        int rp = rowbase + tid;
        sidRow[tid] = (rp < count) ? sidList[e * BB + rp] : -1;
    }
    {   // stage gathered A rows (2 threads per row)
        int i = tid >> 1, hf = tid & 1;
        int rp = rowbase + i;
        uint4* dst = (uint4*)(As + i * AP1 + hf * 512);
        if (rp < count) {
            int sid = sidList[e * BB + rp];
            const uint4* src = (const uint4*)(xb + (size_t)(sid >> 1) * DD + hf * 512);
#pragma unroll
            for (int j = 0; j < 32; ++j) dst[j] = src[j];
        } else {
            uint4 z = make_uint4(0, 0, 0, 0);
#pragma unroll
            for (int j = 0; j < 32; ++j) dst[j] = z;
        }
    }
    const __bf16* w1e = w1t + (size_t)e * HH * DD;
    const int w = tid >> 5, lane = tid & 31;
    const int wm = w >> 1, wn = w & 1;
    const int lh = lane >> 4, l15 = lane & 15;
    v8f acc[2][4];
#pragma unroll
    for (int mt = 0; mt < 2; ++mt)
#pragma unroll
        for (int nt = 0; nt < 4; ++nt)
#pragma unroll
            for (int r = 0; r < 8; ++r) acc[mt][nt][r] = 0.f;

    const int NC = 512;   // chunk c: nb=(c>>4)*128, kk=(c&15)*64
#if HAVE_TDM
    if (tid == 0)
        tdm_load_2d(w1e, (unsigned)(size_t)Bs, 64, 128, DD, TDM_FLAGS_G1);
#endif
    __syncthreads();
    for (int c = 0; c < NC; ++c) {
        const int buf = c & 1;
#if HAVE_TDM
        if (tid == 0) {
            if (c + 1 < NC) {
                int nb1 = ((c + 1) >> 4) * 128, kk1 = ((c + 1) & 15) * 64;
                tdm_load_2d(w1e + (size_t)nb1 * DD + kk1,
                            (unsigned)(size_t)(Bs + (buf ^ 1) * B1SZ), 64, 128, DD, TDM_FLAGS_G1);
                __builtin_amdgcn_s_wait_tensorcnt(1);   // chunk c landed, c+1 in flight
            } else {
                __builtin_amdgcn_s_wait_tensorcnt(0);
            }
        }
        __syncthreads();
#else
        {   // cooperative B staging
            int nb0 = (c >> 4) * 128, kk0 = (c & 15) * 64;
            int i = tid >> 1, hf = tid & 1;
            uint4* dst = (uint4*)(Bs + buf * B1SZ + i * BP1 + hf * 32);
            const uint4* src = (const uint4*)(w1e + (size_t)(nb0 + i) * DD + kk0 + hf * 32);
#pragma unroll
            for (int j = 0; j < 4; ++j) dst[j] = src[j];
        }
        __syncthreads();
#endif
        const int kkb = (c & 15) * 64;
#pragma unroll
        for (int ks = 0; ks < 2; ++ks) {
            const int kl = ks * 32;
            v16bf a[2];
#pragma unroll
            for (int mt = 0; mt < 2; ++mt) {
                const __bf16* p = As + (wm * 32 + mt * 16 + l15) * AP1 + kkb + kl + lh * 8;
                a[mt] = ld16(p, p + 16);
            }
#pragma unroll
            for (int nt = 0; nt < 4; ++nt) {
                const __bf16* bp = Bs + buf * B1SZ + (wn * 64 + nt * 16 + l15) * BP1 + kl + lh * 8;
                v16bf bf = ld16(bp, bp + 16);
#pragma unroll
                for (int mt = 0; mt < 2; ++mt)
                    acc[mt][nt] = __builtin_amdgcn_wmma_f32_16x16x32_bf16(
                        false, a[mt], false, bf, (short)0, acc[mt][nt], false, false);
            }
        }
        if ((c & 15) == 15) {   // finished K for this nb: epilogue
            const int nb = (c >> 4) * 128;
#pragma unroll
            for (int nt = 0; nt < 4; ++nt) {
                int n = nb + wn * 64 + nt * 16 + l15;
                float b1v = eb1[e * HH + n];
#pragma unroll
                for (int mt = 0; mt < 2; ++mt)
#pragma unroll
                    for (int r = 0; r < 8; ++r) {
                        int m = wm * 32 + mt * 16 + lh * 8 + r;
                        int sid = sidRow[m];
                        if (sid >= 0)
                            hbuf[(size_t)sid * HH + n] = f2bf(tanhf(acc[mt][nt][r] + b1v));
                        acc[mt][nt][r] = 0.f;
                    }
            }
        }
        __syncthreads();   // all done reading Bs[buf] before it is overwritten
    }
}

// ================= expert GEMM2: y[sid] = wgt * (h @ W2 + b2), fp32 =================
// block: 256 thr (8 waves, 2M x 4N), M-tile 128, N-block 256, K-chunks 128.
#define AP2 136
#define A2SZ (128 * AP2)
#define BP2 136
#define B2SZ (256 * BP2)
__global__ __launch_bounds__(256) void egemm2_kernel(
    const __bf16* __restrict__ hbuf, const __bf16* __restrict__ w2t,
    const float* __restrict__ eb2, const int* __restrict__ sidList,
    const float* __restrict__ wList, const int* __restrict__ counts,
    float* __restrict__ ybuf) {
    extern __shared__ __align__(16) char smem[];
    __bf16* As = (__bf16*)smem;                 // [2][128][136]
    __bf16* Bs = As + 2 * A2SZ;                 // [2][256][136]
    int* sidRow = (int*)(Bs + 2 * B2SZ);        // 128
    float* wRow = (float*)(sidRow + 128);       // 128
    const int e = blockIdx.y;
    const int count = counts[e];
    const int rowbase = blockIdx.x * 128;
    if (rowbase >= count) return;
    const int tid = threadIdx.x;
    if (tid < 128) {
        int rp = rowbase + tid;
        sidRow[tid] = (rp < count) ? sidList[e * BB + rp] : -1;
        wRow[tid] = (rp < count) ? wList[e * BB + rp] : 0.f;
    }
    __syncthreads();   // sidRow visible for staging
    const __bf16* w2e = w2t + (size_t)e * DD * HH;
    const int w = tid >> 5, lane = tid & 31;
    const int wm = w >> 2, wn = w & 3;
    const int lh = lane >> 4, l15 = lane & 15;
    const int NC = 128;   // chunk cc: nb=(cc>>5)*256, kk=(cc&31)*128

#if HAVE_ASYNC
    {   // zero invalid rows in both A buffers once; async never touches them
        int i = tid >> 1, hf = tid & 1;
        if (sidRow[i] < 0) {
            uint4 z = make_uint4(0, 0, 0, 0);
            uint4* d0 = (uint4*)(As + i * AP2 + hf * 64);
            uint4* d1 = (uint4*)(As + A2SZ + i * AP2 + hf * 64);
#pragma unroll
            for (int j = 0; j < 8; ++j) { d0[j] = z; d1[j] = z; }
        }
    }
    {   // prefetch A(0) async into As[0]
        int i = tid >> 1, hf = tid & 1;
        int sid = sidRow[i];
        if (sid >= 0) {
            const __bf16* src = hbuf + (size_t)sid * HH + hf * 64;
            __bf16* dst = As + i * AP2 + hf * 64;
#pragma unroll
            for (int j = 0; j < 8; ++j) async_b128(src + j * 8, dst + j * 8);
        }
    }
#endif
#if HAVE_TDM
    if (tid == 0)
        tdm_load_2d(w2e, (unsigned)(size_t)Bs, 128, 256, HH, TDM_FLAGS_G2);
#endif
#if HAVE_ASYNC
    __builtin_amdgcn_s_wait_asynccnt(0);
#endif
#if HAVE_TDM
    if (tid == 0) __builtin_amdgcn_s_wait_tensorcnt(0);
#endif
    __syncthreads();

    v8f acc[4][4];
#pragma unroll
    for (int mt = 0; mt < 4; ++mt)
#pragma unroll
        for (int nt = 0; nt < 4; ++nt)
#pragma unroll
            for (int r = 0; r < 8; ++r) acc[mt][nt][r] = 0.f;

    for (int cc = 0; cc < NC; ++cc) {
        const int buf = cc & 1;
        const int kk = (cc & 31) * 128;
#if HAVE_ASYNC
        if (cc + 1 < NC) {   // prefetch A(cc+1) -> As[buf^1] (overlaps WMMA below)
            int kk1 = ((cc + 1) & 31) * 128;
            int i = tid >> 1, hf = tid & 1;
            int sid = sidRow[i];
            if (sid >= 0) {
                const __bf16* src = hbuf + (size_t)sid * HH + kk1 + hf * 64;
                __bf16* dst = As + (buf ^ 1) * A2SZ + i * AP2 + hf * 64;
#pragma unroll
                for (int j = 0; j < 8; ++j) async_b128(src + j * 8, dst + j * 8);
            }
        }
#else
        {   // synchronous A(cc) stage into As[buf]
            int i = tid >> 1, hf = tid & 1;
            int sid = sidRow[i];
            uint4* dst = (uint4*)(As + buf * A2SZ + i * AP2 + hf * 64);
            if (sid >= 0) {
                const uint4* src = (const uint4*)(hbuf + (size_t)sid * HH + kk + hf * 64);
#pragma unroll
                for (int j = 0; j < 8; ++j) dst[j] = src[j];
            } else {
                uint4 z = make_uint4(0, 0, 0, 0);
#pragma unroll
                for (int j = 0; j < 8; ++j) dst[j] = z;
            }
        }
#endif
#if HAVE_TDM
        if (tid == 0) {
            if (cc + 1 < NC) {
                int nb1 = ((cc + 1) >> 5) * 256, kk1 = ((cc + 1) & 31) * 128;
                tdm_load_2d(w2e + (size_t)nb1 * HH + kk1,
                            (unsigned)(size_t)(Bs + (buf ^ 1) * B2SZ), 128, 256, HH, TDM_FLAGS_G2);
                __builtin_amdgcn_s_wait_tensorcnt(1);
            } else {
                __builtin_amdgcn_s_wait_tensorcnt(0);
            }
        }
#else
        {   // cooperative B(cc) stage: 256 rows x 64 elems, one row per thread
            int nb0 = (cc >> 5) * 256;
            uint4* dst = (uint4*)(Bs + buf * B2SZ + tid * BP2);
            const uint4* src = (const uint4*)(w2e + (size_t)(nb0 + tid) * HH + kk);
#pragma unroll
            for (int j = 0; j < 8; ++j) dst[j] = src[j];
        }
#endif
#if HAVE_ASYNC
        // A(cc) is the oldest 8 asyncs; A(cc+1)'s 8 may stay in flight (in-order retire)
        if (cc + 1 < NC) __builtin_amdgcn_s_wait_asynccnt(8);
        else __builtin_amdgcn_s_wait_asynccnt(0);
#endif
        __syncthreads();    // staged data visible to all waves
#pragma unroll
        for (int ks = 0; ks < 4; ++ks) {
            const int kl = ks * 32;
            v16bf a[4];
#pragma unroll
            for (int mt = 0; mt < 4; ++mt) {
                const __bf16* p = As + buf * A2SZ + (wm * 64 + mt * 16 + l15) * AP2 + kl + lh * 8;
                a[mt] = ld16(p, p + 16);
            }
#pragma unroll
            for (int nt = 0; nt < 4; ++nt) {
                const __bf16* bp = Bs + buf * B2SZ + (wn * 64 + nt * 16 + l15) * BP2 + kl + lh * 8;
                v16bf bf = ld16(bp, bp + 16);
#pragma unroll
                for (int mt = 0; mt < 4; ++mt)
                    acc[mt][nt] = __builtin_amdgcn_wmma_f32_16x16x32_bf16(
                        false, a[mt], false, bf, (short)0, acc[mt][nt], false, false);
            }
        }
        if ((cc & 31) == 31) {
            const int nb = (cc >> 5) * 256;
#pragma unroll
            for (int nt = 0; nt < 4; ++nt) {
                int n = nb + wn * 64 + nt * 16 + l15;
                float b2v = eb2[e * DD + n];
#pragma unroll
                for (int mt = 0; mt < 4; ++mt)
#pragma unroll
                    for (int r = 0; r < 8; ++r) {
                        int m = wm * 64 + mt * 16 + lh * 8 + r;
                        int sid = sidRow[m];
                        if (sid >= 0)
                            ybuf[(size_t)sid * DD + n] = wRow[m] * (acc[mt][nt][r] + b2v);
                        acc[mt][nt][r] = 0.f;
                    }
            }
        }
        __syncthreads();    // protect both staging buffers before next overwrite
    }
}

// ---------------- K8: combine the two slots per token ----------------
__global__ void combine_kernel(const float* __restrict__ ybuf, float* __restrict__ out) {
    int idx = blockIdx.x * blockDim.x + threadIdx.x;
    int t = idx >> 10, d = idx & 1023;
    out[idx] = ybuf[(size_t)(2 * t) * DD + d] + ybuf[(size_t)(2 * t + 1) * DD + d];
}

// ---------------- host launcher ----------------
static inline size_t align256(size_t x) { return (x + 255) & ~(size_t)255; }

extern "C" void kernel_launch(void* const* d_in, const int* in_sizes, int n_in,
                              void* d_out, int out_size, void* d_ws, size_t ws_size,
                              hipStream_t stream) {
    (void)in_sizes; (void)n_in; (void)out_size; (void)ws_size;
    const float* x   = (const float*)d_in[0];
    const float* gw1 = (const float*)d_in[1];
    const float* gb1 = (const float*)d_in[2];
    const float* gw2 = (const float*)d_in[3];
    const float* gb2 = (const float*)d_in[4];
    const float* ew1 = (const float*)d_in[5];
    const float* eb1 = (const float*)d_in[6];
    const float* ew2 = (const float*)d_in[7];
    const float* eb2 = (const float*)d_in[8];

    float* out = (float*)d_out;
    int* topk_out = (int*)((float*)d_out + (size_t)BB * DD);

    char* ws = (char*)d_ws;
    size_t o = 0;
    __bf16* xb    = (__bf16*)(ws + o); o = align256(o + (size_t)BB * DD * 2);
    __bf16* gw1t  = (__bf16*)(ws + o); o = align256(o + (size_t)GHH * DD * 2);
    __bf16* w1t   = (__bf16*)(ws + o); o = align256(o + (size_t)EE * HH * DD * 2);
    __bf16* w2t   = (__bf16*)(ws + o); o = align256(o + (size_t)EE * DD * HH * 2);
    __bf16* gh    = (__bf16*)(ws + o); o = align256(o + (size_t)BB * GHH * 2);
    float*  wnorm = (float*)(ws + o);  o = align256(o + (size_t)BB * 2 * 4);
    int*    sidL  = (int*)(ws + o);    o = align256(o + (size_t)EE * BB * 4);
    float*  wL    = (float*)(ws + o);  o = align256(o + (size_t)EE * BB * 4);
    int*    counts= (int*)(ws + o);    o = align256(o + 256);
    __bf16* hbuf  = (__bf16*)(ws + o); o = align256(o + (size_t)BB * KSEL * HH * 2);
    float*  ybuf  = (float*)(ws + o);  o = align256(o + (size_t)BB * KSEL * DD * 4);

    const int smem_g1 = 32 * GP * 2;
    const int smem_e1 = (128 * AP1 + 2 * B1SZ) * 2 + 128 * 4;
    const int smem_e2 = (2 * A2SZ + 2 * B2SZ) * 2 + 128 * 8;
    (void)hipFuncSetAttribute((const void*)gating_gemm1_kernel,
                              hipFuncAttributeMaxDynamicSharedMemorySize, smem_g1);
    (void)hipFuncSetAttribute((const void*)egemm1_kernel,
                              hipFuncAttributeMaxDynamicSharedMemorySize, smem_e1);
    (void)hipFuncSetAttribute((const void*)egemm2_kernel,
                              hipFuncAttributeMaxDynamicSharedMemorySize, smem_e2);

    cvt_x_kernel<<<(BB * DD / 4 + 255) / 256, 256, 0, stream>>>(x, xb, BB * DD / 4);
    transpose_cvt_kernel<<<dim3(GHH / 32, DD / 32, 1), dim3(32, 8), 0, stream>>>(gw1, gw1t, DD, GHH);
    transpose_cvt_kernel<<<dim3(HH / 32, DD / 32, EE), dim3(32, 8), 0, stream>>>(ew1, w1t, DD, HH);
    transpose_cvt_kernel<<<dim3(DD / 32, HH / 32, EE), dim3(32, 8), 0, stream>>>(ew2, w2t, HH, DD);
    gating_gemm1_kernel<<<BB / 32, 256, smem_g1, stream>>>(xb, gw1t, gb1, gh);
    gating2_kernel<<<BB / 32, 256, 0, stream>>>(gh, gw2, gb2, topk_out, wnorm);
    dispatch_kernel<<<EE, 256, 0, stream>>>(topk_out, wnorm, sidL, wL, counts);
    egemm1_kernel<<<dim3(BB / 128, EE), 256, smem_e1, stream>>>(xb, w1t, eb1, sidL, counts, hbuf);
    egemm2_kernel<<<dim3(BB / 128, EE), 256, smem_e2, stream>>>(hbuf, w2t, eb2, sidL, wL, counts, ybuf);
    combine_kernel<<<BB * DD / 256, 256, 0, stream>>>(ybuf, out);
}
